// STMDecoder_79517024518984
// MI455X (gfx1250) — matrix operand order repaired and checked by reading, not compiled, not run on hardware
//
#include <hip/hip_runtime.h>

// ---------------------------------------------------------------------------
// STMDecoder stage for MI455X (gfx1250). All large GEMMs use
// v_wmma_f32_16x16x32_bf16 (wave32), 16x64 tile per wave (4 accumulators,
// A-fragment reused across 4 WMMAs), branch-free software-pipelined K loop
// (double-buffered fragments; WMMAs wait only on previous-iteration loads).
// Weights transposed+converted to bf16 once per launch. f32 accumulation.
// ---------------------------------------------------------------------------

typedef __attribute__((ext_vector_type(16))) __bf16 v16bf;
typedef __attribute__((ext_vector_type(8)))  __bf16 v8bf;
typedef __attribute__((ext_vector_type(8)))  float  v8f;

static constexpr int kB   = 8;
static constexpr int kN   = 300;
static constexpr int kD   = 256;
static constexpr int kNH  = 8;
static constexpr int kG   = 4;
static constexpr int kP   = 32;
static constexpr int kOUT = 128;
static constexpr int kFFN = 2048;
static constexpr int kEFF = 64;     // D / G
static constexpr int kBN  = kB * kN;              // 2400
static constexpr int kPPW = kEFF*kEFF + kP*kOUT;  // 8192 per (b,n,g)
static constexpr int kMIX = kG * kOUT * kEFF;     // 32768 per (b,n)

// ---------------------------------------------------------------------------
// Weight transpose + f32->bf16:  W[K][N] (row major) -> Wt[N][K] bf16
// ---------------------------------------------------------------------------
__global__ void transpose_bf16_kernel(const float* __restrict__ W,
                                      __bf16* __restrict__ Wt, int K, int N) {
  size_t i = (size_t)blockIdx.x * blockDim.x + threadIdx.x;
  if (i >= (size_t)K * N) return;
  int k = (int)(i / N), n = (int)(i % N);
  Wt[(size_t)n * K + k] = (__bf16)W[i];
}

__global__ void tobf16_kernel(const float* __restrict__ a,
                              __bf16* __restrict__ o, size_t n) {
  size_t i = (size_t)blockIdx.x * blockDim.x + threadIdx.x;
  if (i < n) o[i] = (__bf16)a[i];
}

// ---------------------------------------------------------------------------
// Generic WMMA GEMM: C[M][N] = A[M][K] * Bt[N][K]^T + bias, optional ReLU.
// 256 threads = 8 waves per block; each wave owns a 16x64 output tile.
// Two live base pointers only (per-lane chunk offsets folded in; the 4 B
// columns are reached via a wave-uniform SGPR stride 16*K). Unconditional
// pipelined body + WMMA-only epilogue.
// ---------------------------------------------------------------------------
__global__ void gemm_wmma_kernel(const __bf16* __restrict__ A,
                                 const __bf16* __restrict__ Bt,
                                 const float*  __restrict__ bias,
                                 float*  __restrict__ Cf,
                                 __bf16* __restrict__ Cb,
                                 int M, int N, int K, int relu, int nwt) {
  int wt = blockIdx.x * 8 + (threadIdx.x >> 5);   // wave-tile id (wave-uniform)
  if (wt >= nwt) return;
  int ntn = N >> 6;                               // 64-wide column tiles
  int tm = wt / ntn, tn = wt % ntn;
  int lane = threadIdx.x & 31;
  int half = lane >> 4;
  int l16  = lane & 15;
  // fold per-lane K-chunk bases into the pointers
  const __bf16* arow = A  + (size_t)(tm * 16 + l16) * K + half * 16;
  const __bf16* bcol = Bt + (size_t)(tn * 64 + l16) * K + half * 16;
  const size_t cstep = (size_t)16 * K;            // next 16-column group of Bt

  // one warm-up prefetch round (global_prefetch_b8), then rely on pipelining
  __builtin_prefetch(arow, 0, 1);
  __builtin_prefetch(bcol, 0, 1);
  __builtin_prefetch(bcol + cstep, 0, 1);
  __builtin_prefetch(bcol + 2 * cstep, 0, 1);
  __builtin_prefetch(bcol + 3 * cstep, 0, 1);

  v8f acc0 = {}, acc1 = {}, acc2 = {}, acc3 = {};

  // preload fragments for k0 = 0
  v8bf  alo = *(const v8bf*)(arow);
  v8bf  ahi = *(const v8bf*)(arow + 8);
  v16bf nb0 = *(const v16bf*)(bcol);
  v16bf nb1 = *(const v16bf*)(bcol + cstep);
  v16bf nb2 = *(const v16bf*)(bcol + 2 * cstep);
  v16bf nb3 = *(const v16bf*)(bcol + 3 * cstep);

  int k0 = 0;
  for (; k0 + 32 < K; k0 += 32) {
    // materialize current fragments
    v16bf a;
#pragma unroll
    for (int i = 0; i < 8; ++i) { a[i] = alo[i]; a[8 + i] = ahi[i]; }
    v16bf b0 = nb0, b1 = nb1, b2 = nb2, b3 = nb3;

    // issue next-iteration loads before consuming current fragments
    int kn = k0 + 32;
    alo = *(const v8bf*)(arow + kn);
    ahi = *(const v8bf*)(arow + kn + 8);
    nb0 = *(const v16bf*)(bcol + kn);
    nb1 = *(const v16bf*)(bcol + cstep + kn);
    nb2 = *(const v16bf*)(bcol + 2 * cstep + kn);
    nb3 = *(const v16bf*)(bcol + 3 * cstep + kn);

    acc0 = __builtin_amdgcn_wmma_f32_16x16x32_bf16(false, a, false, b0,
                                                   (short)0, acc0, false, false);
    acc1 = __builtin_amdgcn_wmma_f32_16x16x32_bf16(false, a, false, b1,
                                                   (short)0, acc1, false, false);
    acc2 = __builtin_amdgcn_wmma_f32_16x16x32_bf16(false, a, false, b2,
                                                   (short)0, acc2, false, false);
    acc3 = __builtin_amdgcn_wmma_f32_16x16x32_bf16(false, a, false, b3,
                                                   (short)0, acc3, false, false);
  }
  // epilogue: last K-chunk, WMMA only
  {
    v16bf a;
#pragma unroll
    for (int i = 0; i < 8; ++i) { a[i] = alo[i]; a[8 + i] = ahi[i]; }
    acc0 = __builtin_amdgcn_wmma_f32_16x16x32_bf16(false, a, false, nb0,
                                                   (short)0, acc0, false, false);
    acc1 = __builtin_amdgcn_wmma_f32_16x16x32_bf16(false, a, false, nb1,
                                                   (short)0, acc1, false, false);
    acc2 = __builtin_amdgcn_wmma_f32_16x16x32_bf16(false, a, false, nb2,
                                                   (short)0, acc2, false, false);
    acc3 = __builtin_amdgcn_wmma_f32_16x16x32_bf16(false, a, false, nb3,
                                                   (short)0, acc3, false, false);
  }

  v8f accs[4] = { acc0, acc1, acc2, acc3 };
#pragma unroll
  for (int c = 0; c < 4; ++c) {
#pragma unroll
    for (int r = 0; r < 8; ++r) {
      int orow = tm * 16 + r + half * 8;
      int ocol = tn * 64 + c * 16 + l16;
      float v = accs[c][r];
      if (bias) v += bias[ocol];
      if (relu) v = fmaxf(v, 0.0f);
      size_t idx = (size_t)orow * N + ocol;
      if (Cf) Cf[idx] = v;
      if (Cb) Cb[idx] = (__bf16)v;
    }
  }
}

// ---------------------------------------------------------------------------
// Per-row prep: pos-embed, xa = x + pos, box decode. One block per (b,n).
// ---------------------------------------------------------------------------
__global__ void prep_kernel(const float* __restrict__ xyzr,
                            const float* __restrict__ x,
                            float* __restrict__ xa, __bf16* __restrict__ xab,
                            float* __restrict__ boxes, float* __restrict__ wh) {
  size_t bn = blockIdx.x;
  int d = threadIdx.x;                 // 0..255
  const float* q = xyzr + bn * 4;
  int j = d >> 6;                      // coordinate 0..3
  int pair = (d >> 1) & 31;            // frequency index 0..31
  float scale = (j < 2) ? 1000.0f : 1.0f;
  float xv = q[j] / scale;
  float dimt = __powf(10000.0f, (float)pair / 32.0f);
  float ang = xv / dimt;
  float pe = (d & 1) ? __cosf(ang) : __sinf(ang);
  float v = x[bn * kD + d] + pe;
  xa[bn * kD + d] = v;
  xab[bn * kD + d] = (__bf16)v;
  if (d == 0) {
    float sc = exp2f(q[2]);
    float W = sc * exp2f(-0.5f * q[3]);
    float H = sc * exp2f( 0.5f * q[3]);
    wh[bn * 2 + 0] = W; wh[bn * 2 + 1] = H;
    boxes[bn * 4 + 0] = q[0] - 0.5f * W;
    boxes[bn * 4 + 1] = q[1] - 0.5f * H;
    boxes[bn * 4 + 2] = q[0] + 0.5f * W;
    boxes[bn * 4 + 3] = q[1] + 0.5f * H;
  }
}

// ---------------------------------------------------------------------------
// log(IoF + eps) pairwise. grid: (ceil(N*N/256), B)
// ---------------------------------------------------------------------------
__global__ void iof_kernel(const float* __restrict__ boxes,
                           float* __restrict__ ioflog) {
  int b = blockIdx.y;
  int i = blockIdx.x * blockDim.x + threadIdx.x;
  if (i >= kN * kN) return;
  int qq = i / kN, kk = i % kN;
  const float* bq = boxes + ((size_t)b * kN + qq) * 4;
  const float* bk = boxes + ((size_t)b * kN + kk) * 4;
  float lx = fmaxf(bq[0], bk[0]), ly = fmaxf(bq[1], bk[1]);
  float rx = fminf(bq[2], bk[2]), ry = fminf(bq[3], bk[3]);
  float inter = fmaxf(rx - lx, 0.f) * fmaxf(ry - ly, 0.f);
  float area = (bq[2] - bq[0]) * (bq[3] - bq[1]);
  ioflog[((size_t)b * kN + qq) * kN + kk] = __logf(inter / (area + 1e-7f) + 1e-7f);
}

// ---------------------------------------------------------------------------
// Attention: one block per (b,h,q), 128 threads. qkv layout (BN, 768).
// ---------------------------------------------------------------------------
__global__ void attn_kernel(const float* __restrict__ qkv,
                            const float* __restrict__ ioflog,
                            const float* __restrict__ tau,
                            float* __restrict__ o) {
  __shared__ float sc[kN];
  __shared__ float qv[32];
  __shared__ float red[128];
  int idx = blockIdx.x;
  int q = idx % kN; int bh = idx / kN;
  int h = bh % kNH; int b = bh / kNH;
  int t = threadIdx.x;
  const float* qrow = qkv + ((size_t)(b * kN + q)) * (3 * kD) + h * 32;
  if (t < 32) qv[t] = qrow[t];
  __syncthreads();
  float ta = tau[h];
  const float* base = qkv + (size_t)b * kN * (3 * kD);
  const float* bl = ioflog + ((size_t)b * kN + q) * kN;
  for (int k = t; k < kN; k += 128) {
    const float* krow = base + (size_t)k * (3 * kD) + kD + h * 32;
    float s = 0.f;
#pragma unroll
    for (int d = 0; d < 32; ++d) s += qv[d] * krow[d];
    sc[k] = s * 0.17677669529663687f + ta * bl[k];
  }
  __syncthreads();
  float mx = -1e30f;
  for (int k = t; k < kN; k += 128) mx = fmaxf(mx, sc[k]);
  red[t] = mx; __syncthreads();
  for (int s = 64; s > 0; s >>= 1) {
    if (t < s) red[t] = fmaxf(red[t], red[t + s]);
    __syncthreads();
  }
  mx = red[0]; __syncthreads();
  float sm = 0.f;
  for (int k = t; k < kN; k += 128) { float e = __expf(sc[k] - mx); sc[k] = e; sm += e; }
  red[t] = sm; __syncthreads();
  for (int s = 64; s > 0; s >>= 1) {
    if (t < s) red[t] += red[t + s];
    __syncthreads();
  }
  float inv = 1.0f / red[0];
  __syncthreads();
  if (t < 32) {
    float acc = 0.f;
    for (int k = 0; k < kN; ++k)
      acc += sc[k] * base[(size_t)k * (3 * kD) + 2 * kD + h * 32 + t];
    o[((size_t)(b * kN + q)) * kD + h * 32 + t] = acc * inv;
  }
}

// ---------------------------------------------------------------------------
// LayerNorm over D=256 (optional residual, optional ReLU). Block per row.
// ---------------------------------------------------------------------------
__global__ void ln_kernel(const float* __restrict__ a,
                          const float* __restrict__ resid,
                          const float* __restrict__ g,
                          const float* __restrict__ beta,
                          float* __restrict__ outf, __bf16* __restrict__ outb,
                          int relu) {
  __shared__ float red[kD];
  __shared__ float s_mu, s_rs;
  size_t row = blockIdx.x;
  int d = threadIdx.x;
  float v = a[row * kD + d];
  if (resid) v += resid[row * kD + d];
  red[d] = v; __syncthreads();
  for (int s = 128; s > 0; s >>= 1) { if (d < s) red[d] += red[d + s]; __syncthreads(); }
  if (d == 0) s_mu = red[0] / (float)kD;
  __syncthreads();
  float c = v - s_mu;
  red[d] = c * c; __syncthreads();
  for (int s = 128; s > 0; s >>= 1) { if (d < s) red[d] += red[d + s]; __syncthreads(); }
  if (d == 0) s_rs = rsqrtf(red[0] / (float)kD + 1e-5f);
  __syncthreads();
  float ov = c * s_rs * g[d] + beta[d];
  if (relu) ov = fmaxf(ov, 0.f);
  if (outf) outf[row * kD + d] = ov;
  if (outb) outb[row * kD + d] = (__bf16)ov;
}

// ---------------------------------------------------------------------------
// Multi-level bilinear sampling. One block per (b,n,g,p), 64 threads (chan).
// feats are (B, 256, H, W), group g owns channels g*64..g*64+63.
// ---------------------------------------------------------------------------
__global__ void sample_kernel(const float* __restrict__ xyzr,
                              const float* __restrict__ wh,
                              const float* __restrict__ off,
                              const float* __restrict__ f0,
                              const float* __restrict__ f1,
                              const float* __restrict__ f2,
                              const float* __restrict__ f3,
                              __bf16* __restrict__ sampled) {
  int idx = blockIdx.x;
  int p = idx & 31; idx >>= 5;
  int g = idx & 3;  idx >>= 2;
  int n = idx % kN; int b = idx / kN;
  size_t bn = (size_t)b * kN + n;
  const float* orow = off + bn * (kG * kP * 3) + (g * kP + p) * 3;
  float px = xyzr[bn * 4 + 0] + orow[0] * wh[bn * 2 + 0];
  float py = xyzr[bn * 4 + 1] + orow[1] * wh[bn * 2 + 1];
  float lvl = xyzr[bn * 4 + 2] + orow[2] - 3.0f;
  float lw[4], lsum = 0.f;
#pragma unroll
  for (int i = 0; i < 4; ++i) {
    float dd = lvl - (float)i;
    lw[i] = __expf(-0.5f * dd * dd);
    lsum += lw[i];
  }
  float inv = 1.0f / lsum;
  int c = threadIdx.x;
  const float* feats[4] = { f0, f1, f2, f3 };
  const int   dims[4]   = { 128, 64, 32, 16 };
  const float strid[4]  = { 8.f, 16.f, 32.f, 64.f };
  float acc = 0.f;
#pragma unroll
  for (int i = 0; i < 4; ++i) {
    int H = dims[i], W = dims[i];
    float fx = px / strid[i] - 0.5f;
    float fy = py / strid[i] - 0.5f;
    float x0f = floorf(fx), y0f = floorf(fy);
    float ax = fx - x0f, ay = fy - y0f;
    int x0 = (int)x0f, y0 = (int)y0f;
    const float* fp = feats[i] + (((size_t)b * kD + g * kEFF + c) * H) * W;
    float v = 0.f;
#pragma unroll
    for (int dy = 0; dy < 2; ++dy)
#pragma unroll
      for (int dx = 0; dx < 2; ++dx) {
        int xi = x0 + dx, yi = y0 + dy;
        float wgt = (dx ? ax : 1.f - ax) * (dy ? ay : 1.f - ay);
        if (xi >= 0 && xi < W && yi >= 0 && yi < H)
          v += wgt * fp[(size_t)yi * W + xi];
      }
    acc += v * lw[i] * inv;
  }
  sampled[((bn * kG + g) * kP + p) * (size_t)kEFF + c] = (__bf16)acc;
}

// ---------------------------------------------------------------------------
// Adaptive mixer per (b,n,g): C1 = sampled(32x64) @ M(64x64); ln2+relu;
// C2 = S(128x32) @ m1(32x64); ln2+relu -> mixflat bf16. 256 threads (8 waves).
// ---------------------------------------------------------------------------
__global__ void mixer_kernel(const __bf16* __restrict__ sampled,
                             const __bf16* __restrict__ pp,
                             __bf16* __restrict__ mixflat) {
  __shared__ __bf16 sM[64 * 68];   // M matrix, padded row stride
  __shared__ __bf16 sA[32 * 64];   // sampled tile
  __shared__ float  c1[32 * 64];
  __shared__ __bf16 m1[32 * 68];
  __shared__ float  c2[128 * 64];
  __shared__ float  red[256];
  __shared__ float  s_mu, s_rs;

  int bg = blockIdx.x;                       // (b*N+n)*G + g
  int tid = threadIdx.x;
  const __bf16* ppg = pp + (size_t)bg * kPPW;
  const __bf16* Sg  = ppg + kEFF * kEFF;     // 128x32 row-major
  const __bf16* smp = sampled + (size_t)bg * (kP * kEFF);

  for (int i = tid; i < 64 * 64; i += 256) {
    int r = i >> 6, c = i & 63;
    sM[r * 68 + c] = ppg[i];
  }
  for (int i = tid; i < 32 * 64; i += 256) sA[i] = smp[i];
  __syncthreads();

  int wave = tid >> 5, lane = tid & 31;
  int half = lane >> 4, l16 = lane & 15;
  int akb = half * 16, bkb = half * 16;

  // ---- stage A: 2x4 tiles of 16x16, one per wave, K=64 ----
  {
    int tm = wave >> 2, tn = wave & 3;
    int row = tm * 16 + l16;
    int col = tn * 16 + l16;
    v8f acc = {};
    for (int k0 = 0; k0 < 64; k0 += 32) {
      v16bf a, b;
#pragma unroll
      for (int i = 0; i < 8; ++i) {
        a[i]     = sA[row * 64 + k0 + akb + i];
        a[8 + i] = sA[row * 64 + k0 + akb + 8 + i];
      }
#pragma unroll
      for (int i = 0; i < 16; ++i) b[i] = sM[(k0 + bkb + i) * 68 + col];
      acc = __builtin_amdgcn_wmma_f32_16x16x32_bf16(false, a, false, b,
                                                    (short)0, acc, false, false);
    }
#pragma unroll
    for (int r = 0; r < 8; ++r)
      c1[(tm * 16 + r + half * 8) * 64 + tn * 16 + l16] = acc[r];
  }
  __syncthreads();

  // ---- ln2 + relu over c1 (2048) ----
  {
    float p1 = 0.f, p2 = 0.f;
    for (int i = tid; i < 2048; i += 256) { float v = c1[i]; p1 += v; p2 += v * v; }
    red[tid] = p1; __syncthreads();
    for (int s = 128; s > 0; s >>= 1) { if (tid < s) red[tid] += red[tid + s]; __syncthreads(); }
    if (tid == 0) s_mu = red[0] / 2048.0f;
    __syncthreads();
    red[tid] = p2; __syncthreads();
    for (int s = 128; s > 0; s >>= 1) { if (tid < s) red[tid] += red[tid + s]; __syncthreads(); }
    if (tid == 0) s_rs = rsqrtf(red[0] / 2048.0f - s_mu * s_mu + 1e-5f);
    __syncthreads();
    float mu = s_mu, rs = s_rs;
    for (int i = tid; i < 2048; i += 256) {
      float v = fmaxf((c1[i] - mu) * rs, 0.f);
      m1[(i >> 6) * 68 + (i & 63)] = (__bf16)v;
    }
  }
  __syncthreads();

  // ---- stage B: wave w owns 16 rows of S (tm=w); A-frag reused over 4 tn ----
  {
    int tm = wave;
    int rowq = tm * 16 + l16;
    v16bf a;
    v8bf alo = *(const v8bf*)(Sg + rowq * 32 + akb);
    v8bf ahi = *(const v8bf*)(Sg + rowq * 32 + akb + 8);
#pragma unroll
    for (int i = 0; i < 8; ++i) { a[i] = alo[i]; a[8 + i] = ahi[i]; }
    v8f acc[4] = {};
#pragma unroll
    for (int tn = 0; tn < 4; ++tn) {
      v16bf b;
#pragma unroll
      for (int i = 0; i < 16; ++i) b[i] = m1[(bkb + i) * 68 + tn * 16 + l16];
      acc[tn] = __builtin_amdgcn_wmma_f32_16x16x32_bf16(false, a, false, b,
                                                        (short)0, acc[tn], false, false);
    }
#pragma unroll
    for (int tn = 0; tn < 4; ++tn)
#pragma unroll
      for (int r = 0; r < 8; ++r)
        c2[(tm * 16 + r + half * 8) * 64 + tn * 16 + l16] = acc[tn][r];
  }
  __syncthreads();

  // ---- ln2 + relu over c2 (8192), emit bf16 into mixflat ----
  {
    float p1 = 0.f, p2 = 0.f;
    for (int i = tid; i < 8192; i += 256) { float v = c2[i]; p1 += v; p2 += v * v; }
    red[tid] = p1; __syncthreads();
    for (int s = 128; s > 0; s >>= 1) { if (tid < s) red[tid] += red[tid + s]; __syncthreads(); }
    if (tid == 0) s_mu = red[0] / 8192.0f;
    __syncthreads();
    red[tid] = p2; __syncthreads();
    for (int s = 128; s > 0; s >>= 1) { if (tid < s) red[tid] += red[tid + s]; __syncthreads(); }
    if (tid == 0) s_rs = rsqrtf(red[0] / 8192.0f - s_mu * s_mu + 1e-5f);
    __syncthreads();
    float mu = s_mu, rs = s_rs;
    __bf16* outg = mixflat + (size_t)(bg / kG) * kMIX + (size_t)(bg % kG) * (kOUT * kEFF);
    for (int i = tid; i < 8192; i += 256) {
      float v = fmaxf((c2[i] - mu) * rs, 0.f);
      outg[i] = (__bf16)v;
    }
  }
}

// ---------------------------------------------------------------------------
// Final tiny heads: cls (2) and box delta (4) per (b,n). Block of 64 threads.
// ---------------------------------------------------------------------------
__global__ void heads_kernel(const float* __restrict__ c,
                             const float* __restrict__ r,
                             const float* __restrict__ wcls,
                             const float* __restrict__ bcls,
                             const float* __restrict__ wreg,
                             const float* __restrict__ breg,
                             const float* __restrict__ xyzr,
                             float* __restrict__ out_cls,
                             float* __restrict__ out_xyzr) {
  __shared__ float dlt[4];
  size_t bn = blockIdx.x;
  int t = threadIdx.x;
  if (t < 2) {
    float acc = bcls[t];
    for (int d = 0; d < kD; ++d) acc += c[bn * kD + d] * wcls[d * 2 + t];
    out_cls[bn * 2 + t] = acc;
  } else if (t < 6) {
    int j = t - 2;
    float acc = breg[j];
    for (int d = 0; d < kD; ++d) acc += r[bn * kD + d] * wreg[d * 4 + j];
    dlt[j] = acc;
  }
  __syncthreads();
  if (t == 0) {
    const float* q = xyzr + bn * 4;
    float s = exp2f(q[2]);
    out_xyzr[bn * 4 + 0] = q[0] + dlt[0] * s;
    out_xyzr[bn * 4 + 1] = q[1] + dlt[1] * s;
    out_xyzr[bn * 4 + 2] = q[2] + dlt[2];
    out_xyzr[bn * 4 + 3] = q[3] + dlt[3];
  }
}

// ---------------------------------------------------------------------------
// Launch
// ---------------------------------------------------------------------------
static inline void* wsalloc(char*& p, size_t bytes) {
  void* r = (void*)p;
  p += (bytes + 255) & ~(size_t)255;
  return r;
}

static inline void gemm(hipStream_t stream, const __bf16* A, const __bf16* Bt,
                        const float* bias, float* Cf, __bf16* Cb,
                        int M, int N, int K, int relu) {
  int nwt = (M >> 4) * (N >> 6);       // wave-tiles (16x64 each); all N % 64 == 0
  int blocks = (nwt + 7) / 8;
  gemm_wmma_kernel<<<blocks, 256, 0, stream>>>(A, Bt, bias, Cf, Cb, M, N, K, relu, nwt);
}

extern "C" void kernel_launch(void* const* d_in, const int* in_sizes, int n_in,
                              void* d_out, int out_size, void* d_ws, size_t ws_size,
                              hipStream_t stream) {
  // ---- inputs (setup_inputs() dict order; params flattened w,b per layer) ----
  const float* f0   = (const float*)d_in[0];
  const float* f1   = (const float*)d_in[1];
  const float* f2   = (const float*)d_in[2];
  const float* f3   = (const float*)d_in[3];
  const float* xyzr = (const float*)d_in[4];
  const float* xin  = (const float*)d_in[5];
  const float* w_attn_in  = (const float*)d_in[6];   const float* b_attn_in  = (const float*)d_in[7];
  const float* w_attn_out = (const float*)d_in[8];   const float* b_attn_out = (const float*)d_in[9];
  const float* g_attn_ln  = (const float*)d_in[10];  const float* b_attn_ln  = (const float*)d_in[11];
  const float* iof_tau    = (const float*)d_in[12];
  const float* w_off      = (const float*)d_in[13];  const float* b_off      = (const float*)d_in[14];
  const float* w_mixp     = (const float*)d_in[15];  const float* b_mixp     = (const float*)d_in[16];
  const float* w_mixo     = (const float*)d_in[17];  const float* b_mixo     = (const float*)d_in[18];
  const float* g_norm_s   = (const float*)d_in[19];  const float* b_norm_s   = (const float*)d_in[20];
  const float* w_ffn1     = (const float*)d_in[21];  const float* b_ffn1     = (const float*)d_in[22];
  const float* w_ffn2     = (const float*)d_in[23];  const float* b_ffn2     = (const float*)d_in[24];
  const float* g_ffn_ln   = (const float*)d_in[25];  const float* b_ffn_ln   = (const float*)d_in[26];
  const float* w_cls_fc   = (const float*)d_in[27];  const float* b_cls_fc   = (const float*)d_in[28];
  const float* g_cls_ln   = (const float*)d_in[29];  const float* b_cls_ln   = (const float*)d_in[30];
  const float* w_fc_cls   = (const float*)d_in[31];  const float* b_fc_cls   = (const float*)d_in[32];
  const float* w_reg_fc   = (const float*)d_in[33];  const float* b_reg_fc   = (const float*)d_in[34];
  const float* g_reg_ln   = (const float*)d_in[35];  const float* b_reg_ln   = (const float*)d_in[36];
  const float* w_fc_reg   = (const float*)d_in[37];  const float* b_fc_reg   = (const float*)d_in[38];

  // ---- outputs: (cls, new_xyzr, x) concatenated flat, f32 ----
  float* out_cls  = (float*)d_out;                       // 8*300*2
  float* out_xyzr = out_cls + kBN * 2;                   // 8*300*4
  float* out_x    = out_xyzr + kBN * 4;                  // 8*300*256

  // ---- workspace carve-up ----
  char* wp = (char*)d_ws;
  __bf16* t_attn_in  = (__bf16*)wsalloc(wp, (size_t)768 * kD * 2);
  __bf16* t_attn_out = (__bf16*)wsalloc(wp, (size_t)kD * kD * 2);
  __bf16* t_off      = (__bf16*)wsalloc(wp, (size_t)384 * kD * 2);
  __bf16* t_mixp     = (__bf16*)wsalloc(wp, (size_t)(kG * kPPW) * kD * 2);
  __bf16* t_mixo     = (__bf16*)wsalloc(wp, (size_t)kD * kMIX * 2);
  __bf16* t_ffn1     = (__bf16*)wsalloc(wp, (size_t)kFFN * kD * 2);
  __bf16* t_ffn2     = (__bf16*)wsalloc(wp, (size_t)kD * kFFN * 2);
  __bf16* t_cls_fc   = (__bf16*)wsalloc(wp, (size_t)kD * kD * 2);
  __bf16* t_reg_fc   = (__bf16*)wsalloc(wp, (size_t)kD * kD * 2);

  float*  xa     = (float*) wsalloc(wp, (size_t)kBN * kD * 4);
  __bf16* xab    = (__bf16*)wsalloc(wp, (size_t)kBN * kD * 2);
  float*  boxes  = (float*) wsalloc(wp, (size_t)kBN * 4 * 4);
  float*  whb    = (float*) wsalloc(wp, (size_t)kBN * 2 * 4);
  float*  ioflog = (float*) wsalloc(wp, (size_t)kB * kN * kN * 4);
  float*  qkv    = (float*) wsalloc(wp, (size_t)kBN * 3 * kD * 4);
  float*  obuf   = (float*) wsalloc(wp, (size_t)kBN * kD * 4);
  __bf16* obufb  = (__bf16*)wsalloc(wp, (size_t)kBN * kD * 2);
  float*  y1     = (float*) wsalloc(wp, (size_t)kBN * kD * 4);
  float*  x1     = (float*) wsalloc(wp, (size_t)kBN * kD * 4);
  __bf16* x1b    = (__bf16*)wsalloc(wp, (size_t)kBN * kD * 2);
  float*  offb   = (float*) wsalloc(wp, (size_t)kBN * kG * kP * 3 * 4);
  __bf16* smp    = (__bf16*)wsalloc(wp, (size_t)kBN * kG * kP * kEFF * 2);
  __bf16* ppb    = (__bf16*)wsalloc(wp, (size_t)kBN * kG * kPPW * 2);
  __bf16* mixf   = (__bf16*)wsalloc(wp, (size_t)kBN * kMIX * 2);
  float*  y2     = (float*) wsalloc(wp, (size_t)kBN * kD * 4);
  float*  x2     = (float*) wsalloc(wp, (size_t)kBN * kD * 4);
  __bf16* x2b    = (__bf16*)wsalloc(wp, (size_t)kBN * kD * 2);
  __bf16* hb     = (__bf16*)wsalloc(wp, (size_t)kBN * kFFN * 2);
  float*  y3     = (float*) wsalloc(wp, (size_t)kBN * kD * 4);
  __bf16* x3b    = (__bf16*)wsalloc(wp, (size_t)kBN * kD * 2);
  float*  c1     = (float*) wsalloc(wp, (size_t)kBN * kD * 4);
  float*  cfin   = (float*) wsalloc(wp, (size_t)kBN * kD * 4);
  float*  r1     = (float*) wsalloc(wp, (size_t)kBN * kD * 4);
  float*  rfin   = (float*) wsalloc(wp, (size_t)kBN * kD * 4);
  (void)ws_size; (void)in_sizes; (void)n_in; (void)out_size;

  auto tr = [&](const float* W, __bf16* Wt, int K, int N) {
    size_t cnt = (size_t)K * N;
    transpose_bf16_kernel<<<(unsigned)((cnt + 255) / 256), 256, 0, stream>>>(W, Wt, K, N);
  };

  // ---- weight prep (bf16, N x K layout for B-fragments) ----
  tr(w_attn_in,  t_attn_in,  kD, 3 * kD);
  tr(w_attn_out, t_attn_out, kD, kD);
  tr(w_off,      t_off,      kD, kG * kP * 3);
  tr(w_mixp,     t_mixp,     kD, kG * kPPW);
  tr(w_mixo,     t_mixo,     kMIX, kD);
  tr(w_ffn1,     t_ffn1,     kD, kFFN);
  tr(w_ffn2,     t_ffn2,     kFFN, kD);
  tr(w_cls_fc,   t_cls_fc,   kD, kD);
  tr(w_reg_fc,   t_reg_fc,   kD, kD);

  // ---- stage ----
  prep_kernel<<<kBN, kD, 0, stream>>>(xyzr, xin, xa, xab, boxes, whb);
  {
    dim3 g((kN * kN + 255) / 256, kB);
    iof_kernel<<<g, 256, 0, stream>>>(boxes, ioflog);
  }
  gemm(stream, xab, t_attn_in, b_attn_in, qkv, nullptr, kBN, 3 * kD, kD, 0);
  attn_kernel<<<kB * kNH * kN, 128, 0, stream>>>(qkv, ioflog, iof_tau, obuf);
  tobf16_kernel<<<(kBN * kD + 255) / 256, 256, 0, stream>>>(obuf, obufb, (size_t)kBN * kD);
  gemm(stream, obufb, t_attn_out, b_attn_out, y1, nullptr, kBN, kD, kD, 0);
  ln_kernel<<<kBN, kD, 0, stream>>>(y1, xa, g_attn_ln, b_attn_ln, x1, x1b, 0);

  gemm(stream, x1b, t_off, b_off, offb, nullptr, kBN, kG * kP * 3, kD, 0);
  sample_kernel<<<kBN * kG * kP, kEFF, 0, stream>>>(xyzr, whb, offb, f0, f1, f2, f3, smp);
  gemm(stream, x1b, t_mixp, b_mixp, nullptr, ppb, kBN, kG * kPPW, kD, 0);
  mixer_kernel<<<kBN * kG, 256, 0, stream>>>(smp, ppb, mixf);
  gemm(stream, mixf, t_mixo, b_mixo, y2, nullptr, kBN, kD, kMIX, 0);
  ln_kernel<<<kBN, kD, 0, stream>>>(y2, x1, g_norm_s, b_norm_s, x2, x2b, 0);

  gemm(stream, x2b, t_ffn1, b_ffn1, nullptr, hb, kBN, kFFN, kD, 1);
  gemm(stream, hb, t_ffn2, b_ffn2, y3, nullptr, kBN, kD, kFFN, 0);
  ln_kernel<<<kBN, kD, 0, stream>>>(y3, x2, g_ffn_ln, b_ffn_ln, out_x, x3b, 0);

  gemm(stream, x3b, t_cls_fc, b_cls_fc, c1, nullptr, kBN, kD, kD, 0);
  ln_kernel<<<kBN, kD, 0, stream>>>(c1, nullptr, g_cls_ln, b_cls_ln, cfin, nullptr, 1);
  gemm(stream, x3b, t_reg_fc, b_reg_fc, r1, nullptr, kBN, kD, kD, 0);
  ln_kernel<<<kBN, kD, 0, stream>>>(r1, nullptr, g_reg_ln, b_reg_ln, rfin, nullptr, 1);
  heads_kernel<<<kBN, 64, 0, stream>>>(cfin, rfin, w_fc_cls, b_fc_cls,
                                       w_fc_reg, b_fc_reg, xyzr, out_cls, out_xyzr);
}
